// RPN_48567490183661
// MI455X (gfx1250) — matrix operand order repaired and checked by reading, not compiled, not run on hardware
//
#include <hip/hip_runtime.h>
#include <hip/hip_bf16.h>
#include <math.h>

typedef _Float16 v16h __attribute__((ext_vector_type(16)));
typedef float    v8f  __attribute__((ext_vector_type(8)));
union F16x16 { unsigned u[8]; v16h v; };
union F16Q   { uint4 q[2]; v16h v; };

#define IMG     1600.0f
#define FEATN   10000            // 100x100 spatial
#define CCH     512
#define PADW    102              // padded spatial edge
#define NA      9
#define NPROP   90000            // FEATN * NA
#define NSORT   131072
#define NPRE    2000
#define NPOST   1000
#define BBCLIP  4.135166556742356f
// d_out layout (floats): props[1000*4] | scores[1000] | anchors[90000*4] | cls[90000] | box[90000*4]
#define O_SCORE 4000
#define O_ANCH  5000
#define O_CLS   365000
#define O_BOX   455000

// ---------------- pack kernels ----------------
// feat NCHW f32 -> padded NHWC f16 (102x102 with zero halo)
__global__ void k_pack_feat(const float* __restrict__ f, _Float16* __restrict__ o) {
  int e = blockIdx.x * blockDim.x + threadIdx.x;      // 102*102*512 = 5,326,848
  if (e >= PADW * PADW * CCH) return;
  int c = e % CCH, m = e / CCH;
  int px = m % PADW, py = m / PADW;
  float v = 0.f;
  if (py >= 1 && py <= 100 && px >= 1 && px <= 100)
    v = f[(size_t)c * FEATN + (py - 1) * 100 + (px - 1)];
  o[e] = (_Float16)v;
}
// Pack 3x3 weights directly into WMMA B-fragment order:
//   o[(((tap*16+kc)*32+nt)*32+lane)*16 + e16] = W[oc=nt*16+(lane&15)][ic=kc*32+(lane>>4)*16+e16][ky][kx]
__global__ void k_pack_w3(const float* __restrict__ w, _Float16* __restrict__ o) {
  int e = blockIdx.x * blockDim.x + threadIdx.x;      // 9 * 2^18 = 2,359,296
  if (e >= CCH * CCH * 9) return;
  int e16 = e & 15;
  int l   = (e >> 4) & 31;
  int nt  = (e >> 9) & 31;
  int kc  = (e >> 14) & 15;
  int tap = e >> 18;
  int half = l >> 4, n = l & 15;
  int ic = kc * 32 + half * 16 + e16;
  int oc = nt * 16 + n;
  int ky = tap / 3, kx = tap % 3;
  o[e] = (_Float16)w[(((size_t)oc * CCH + ic) * 3 + ky) * 3 + kx];
}
__global__ void k_pack_w1(const float* __restrict__ cw, const float* __restrict__ rw,
                          _Float16* __restrict__ o) {
  int e = blockIdx.x * blockDim.x + threadIdx.x;      // 48*512
  if (e >= 48 * CCH) return;
  int j = e / CCH, ic = e % CCH;
  float v = 0.f;
  if (j < 9) v = cw[j * CCH + ic];
  else if (j < 45) v = rw[(j - 9) * CCH + ic];
  o[e] = (_Float16)v;
}

// ---------------- 3x3 conv: LDS-free implicit GEMM via WMMA ----------------
// grid 625, block 256 (8 waves). Wave w owns 4 oc-tiles (nt = w*4..w*4+3):
// one A fragment feeds 4 WMMAs -> 4x A reuse, no boundary predication (padded input).
__global__ void __launch_bounds__(256) k_conv3(const _Float16* __restrict__ featP,
                                               const _Float16* __restrict__ bfrag,
                                               const float* __restrict__ bconv,
                                               _Float16* __restrict__ rpnH) {
  const int t = threadIdx.x;
  const int lane = t & 31, w = t >> 5;
  const int Mt = blockIdx.x;                // 625 spatial tiles
  const int row = lane & 15, half = lane >> 4;
  const int nt0 = w * 4;                    // first of 4 oc-tiles for this wave
  const int m  = Mt * 16 + row;
  const int yy = m / 100, xx = m % 100;
  v8f acc[4];
  #pragma unroll
  for (int u = 0; u < 4; ++u) acc[u] = (v8f){0.f,0.f,0.f,0.f,0.f,0.f,0.f,0.f};
  for (int ky = 0; ky < 3; ++ky) {
    for (int kx = 0; kx < 3; ++kx) {
      const int tap = ky * 3 + kx;
      const uint4* arow  = (const uint4*)(featP + (size_t)((yy + ky) * PADW + (xx + kx)) * CCH);
      const uint4* bb    = (const uint4*)bfrag + (size_t)tap * 32768 + (nt0 * 32 + lane) * 2;
      for (int kc = 0; kc < 16; ++kc) {
        F16Q af;
        af.q[0] = arow[kc * 4 + 2 * half];     // f16 [kc*32 + 16*half .. +8)
        af.q[1] = arow[kc * 4 + 2 * half + 1]; // f16 [kc*32 + 16*half + 8 .. +8)
        const uint4* bk = bb + (size_t)kc * 2048;
        __builtin_prefetch((const void*)(bk + 2048), 0, 1);
        F16Q bf0, bf1, bf2, bf3;
        bf0.q[0] = bk[0];   bf0.q[1] = bk[1];     // nt0+0 (nt stride = 64 uint4)
        bf1.q[0] = bk[64];  bf1.q[1] = bk[65];    // nt0+1
        bf2.q[0] = bk[128]; bf2.q[1] = bk[129];   // nt0+2
        bf3.q[0] = bk[192]; bf3.q[1] = bk[193];   // nt0+3
        acc[0] = __builtin_amdgcn_wmma_f32_16x16x32_f16(false, af.v, false, bf0.v,
                                                        (short)0, acc[0], false, false);
        acc[1] = __builtin_amdgcn_wmma_f32_16x16x32_f16(false, af.v, false, bf1.v,
                                                        (short)0, acc[1], false, false);
        acc[2] = __builtin_amdgcn_wmma_f32_16x16x32_f16(false, af.v, false, bf2.v,
                                                        (short)0, acc[2], false, false);
        acc[3] = __builtin_amdgcn_wmma_f32_16x16x32_f16(false, af.v, false, bf3.v,
                                                        (short)0, acc[3], false, false);
      }
    }
  }
  #pragma unroll
  for (int u = 0; u < 4; ++u) {
    const int oc = (nt0 + u) * 16 + row;
    const float bias = bconv[oc];
    #pragma unroll
    for (int r = 0; r < 8; ++r) {
      const int mm = Mt * 16 + r + 8 * half;
      float v = acc[u][r] + bias;
      v = v > 0.f ? v : 0.f;                // ReLU
      rpnH[(size_t)mm * CCH + oc] = (_Float16)v;
    }
  }
}

// ---------------- 1x1 heads: WMMA GEMM M=10000 N=48 K=512 ----------------
__global__ void __launch_bounds__(96) k_conv1(const _Float16* __restrict__ rpnH,
                                              const _Float16* __restrict__ w1h,
                                              const float* __restrict__ cls_b,
                                              const float* __restrict__ reg_b,
                                              float* __restrict__ out,
                                              float* __restrict__ scores) {
  const int t = threadIdx.x, lane = t & 31, w = t >> 5;  // 3 waves, one N-tile each
  const int Mt = blockIdx.x;
  const int row = lane & 15, half = lane >> 4;
  const unsigned* arow = (const unsigned*)(rpnH + (size_t)(Mt * 16 + row) * CCH);
  const int j = w * 16 + row;
  const unsigned* brow = (const unsigned*)(w1h + (size_t)j * CCH);
  v8f acc = {0.f, 0.f, 0.f, 0.f, 0.f, 0.f, 0.f, 0.f};
  for (int kc = 0; kc < 16; ++kc) {
    F16x16 af, bf;
    #pragma unroll
    for (int v = 0; v < 8; ++v)
      af.u[v] = arow[kc * 16 + half * 8 + v];
    #pragma unroll
    for (int k2 = 0; k2 < 8; ++k2)
      bf.u[k2] = brow[kc * 16 + half * 8 + k2];
    acc = __builtin_amdgcn_wmma_f32_16x16x32_f16(false, af.v, false, bf.v,
                                                 (short)0, acc, false, false);
  }
  if (j < 45) {
    const float bias = (j < 9) ? cls_b[j] : reg_b[j - 9];
    #pragma unroll
    for (int r = 0; r < 8; ++r) {
      const int mm = Mt * 16 + r + 8 * half;
      float v = acc[r] + bias;
      if (j < 9) {
        out[O_CLS + (size_t)mm * 9 + j] = v;
        scores[(size_t)mm * 9 + j] = 1.f / (1.f + expf(-v));
      } else {
        out[O_BOX + (size_t)mm * 36 + (j - 9)] = v;
      }
    }
  }
}

// ---------------- anchors + decode ----------------
__global__ void k_anchors(float* __restrict__ out) {
  int i = blockIdx.x * blockDim.x + threadIdx.x;
  if (i >= NPROP) return;
  int m = i / NA, a = i % NA;
  int yy = m / 100, xx = m % 100;
  const float ars[3] = {0.5f, 1.0f, 2.0f};
  const float scs[3] = {128.f, 256.f, 512.f};
  float hr = sqrtf(ars[a / 3]), wr = 1.f / hr;
  float ws = wr * scs[a % 3], hs = hr * scs[a % 3];
  float bx1 = roundf(-ws * 0.5f), by1 = roundf(-hs * 0.5f);
  float bx2 = roundf( ws * 0.5f), by2 = roundf( hs * 0.5f);
  float sx = (float)(xx * 16), sy = (float)(yy * 16);
  out[O_ANCH + (size_t)i * 4 + 0] = sx + bx1;
  out[O_ANCH + (size_t)i * 4 + 1] = sy + by1;
  out[O_ANCH + (size_t)i * 4 + 2] = sx + bx2;
  out[O_ANCH + (size_t)i * 4 + 3] = sy + by2;
}
__global__ void k_decode(const float* __restrict__ out, float* __restrict__ prop) {
  int i = blockIdx.x * blockDim.x + threadIdx.x;
  if (i >= NPROP) return;
  const float* an = out + O_ANCH + (size_t)i * 4;
  const float* de = out + O_BOX  + (size_t)i * 4;
  float w = an[2] - an[0], h = an[3] - an[1];
  float cx = an[0] + 0.5f * w, cy = an[1] + 0.5f * h;
  float dx = de[0], dy = de[1];
  float dw = fminf(de[2], BBCLIP), dh = fminf(de[3], BBCLIP);
  float pcx = dx * w + cx, pcy = dy * h + cy;
  float pw = expf(dw) * w, ph = expf(dh) * h;
  prop[(size_t)i * 4 + 0] = pcx - 0.5f * pw;
  prop[(size_t)i * 4 + 1] = pcy - 0.5f * ph;
  prop[(size_t)i * 4 + 2] = pcx + 0.5f * pw;
  prop[(size_t)i * 4 + 3] = pcy + 0.5f * ph;
}

// ---------------- exact top-k via bitonic sort (desc, tie -> lower idx) ----------------
__global__ void k_sort_init(const float* __restrict__ s, float* __restrict__ key,
                            int* __restrict__ idx) {
  int i = blockIdx.x * blockDim.x + threadIdx.x;
  if (i >= NSORT) return;
  key[i] = (i < NPROP) ? s[i] : -3.0e38f;
  idx[i] = i;
}
__global__ void k_sort_pass(float* __restrict__ key, int* __restrict__ idx, int k, int jj) {
  int i = blockIdx.x * blockDim.x + threadIdx.x;
  int ixj = i ^ jj;
  if (ixj <= i) return;
  float ka = key[i], kb = key[ixj];
  int ia = idx[i], ib = idx[ixj];
  bool up = ((i & k) == 0);
  bool b_first = (kb > ka) || (kb == ka && ib < ia);
  if (up == b_first) { key[i] = kb; key[ixj] = ka; idx[i] = ib; idx[ixj] = ia; }
}

// ---------------- gather/clip/valid for top-2000 ----------------
__global__ void k_gather(const float* __restrict__ prop, const int* __restrict__ sidx,
                         float* __restrict__ p2, unsigned* __restrict__ valid) {
  int i = blockIdx.x * blockDim.x + threadIdx.x;
  if (i >= NPRE) return;
  int s = sidx[i];
  float x1 = fminf(fmaxf(prop[(size_t)s * 4 + 0], 0.f), IMG);
  float y1 = fminf(fmaxf(prop[(size_t)s * 4 + 1], 0.f), IMG);
  float x2 = fminf(fmaxf(prop[(size_t)s * 4 + 2], 0.f), IMG);
  float y2 = fminf(fmaxf(prop[(size_t)s * 4 + 3], 0.f), IMG);
  p2[i * 4 + 0] = x1; p2[i * 4 + 1] = y1; p2[i * 4 + 2] = x2; p2[i * 4 + 3] = y2;
  valid[i] = ((x2 - x1 >= 16.f) && (y2 - y1 >= 16.f)) ? 1u : 0u;
}

// ---------------- NMS: IoU bitmask + sequential scan ----------------
__global__ void k_iou(const float* __restrict__ p, unsigned* __restrict__ mask) {
  int i = blockIdx.x, w = threadIdx.x;
  if (w >= 63) return;
  float x1 = p[i * 4], y1 = p[i * 4 + 1], x2 = p[i * 4 + 2], y2 = p[i * 4 + 3];
  float ai = (x2 - x1) * (y2 - y1);
  unsigned bits = 0;
  for (int b = 0; b < 32; ++b) {
    int j = w * 32 + b;
    if (j >= NPRE) break;
    float u1 = p[j * 4], v1 = p[j * 4 + 1], u2 = p[j * 4 + 2], v2 = p[j * 4 + 3];
    float aj = (u2 - u1) * (v2 - v1);
    float iw = fmaxf(fminf(x2, u2) - fmaxf(x1, u1), 0.f);
    float ih = fmaxf(fminf(y2, v2) - fmaxf(y1, v1), 0.f);
    float inter = iw * ih;
    float un = fmaxf(ai + aj - inter, 1e-9f);
    if (inter / un > 0.7f) bits |= (1u << b);
  }
  mask[(size_t)i * 63 + w] = bits;
}
__global__ void k_nms(const unsigned* __restrict__ mask, const unsigned* __restrict__ valid,
                      unsigned* __restrict__ keep) {
  __shared__ unsigned keepw[63];
  __shared__ unsigned flag;
  int t = threadIdx.x;
  if (t < 63) keepw[t] = 0u;
  if (t == 0) flag = 0u;
  __syncthreads();
  for (int i = 0; i < NPRE; ++i) {
    if (t < 63) {
      unsigned mrow = mask[(size_t)i * 63 + t] & keepw[t];
      int wl = i >> 5, bl = i & 31;
      if (t > wl) mrow = 0u;
      else if (t == wl) mrow &= (bl ? (0xFFFFFFFFu >> (32 - bl)) : 0u);
      if (mrow) atomicOr(&flag, 1u);
    }
    __syncthreads();
    if (t == 0) {
      unsigned ki = (valid[i] && !flag) ? 1u : 0u;
      keep[i] = ki;
      if (ki) keepw[i >> 5] |= (1u << (i & 31));
      flag = 0u;
    }
    __syncthreads();
  }
}
__global__ void k_final(const unsigned* __restrict__ keep, const float* __restrict__ skey,
                        const float* __restrict__ p2, int* __restrict__ order,
                        float* __restrict__ out) {
  int t = threadIdx.x;
  if (t == 0) {
    int T = 0;
    for (int i = 0; i < NPRE; ++i) T += keep[i] ? 1 : 0;
    int a = 0, b = 0;
    for (int i = 0; i < NPRE; ++i) order[i] = keep[i] ? (a++) : (T + b++);
  }
  __syncthreads();
  for (int i = t; i < NPRE; i += blockDim.x) {
    int p = order[i];
    if (p < NPOST) {
      out[O_SCORE + p] = keep[i] ? skey[i] : -1.0f;
      out[p * 4 + 0] = p2[i * 4 + 0];
      out[p * 4 + 1] = p2[i * 4 + 1];
      out[p * 4 + 2] = p2[i * 4 + 2];
      out[p * 4 + 3] = p2[i * 4 + 3];
    }
  }
}

// ---------------- launch ----------------
extern "C" void kernel_launch(void* const* d_in, const int* in_sizes, int n_in,
                              void* d_out, int out_size, void* d_ws, size_t ws_size,
                              hipStream_t stream) {
  (void)in_sizes; (void)n_in; (void)out_size; (void)ws_size;
  const float* feat   = (const float*)d_in[1];
  const float* w3     = (const float*)d_in[2];
  const float* b3     = (const float*)d_in[3];
  const float* cls_w  = (const float*)d_in[4];
  const float* cls_b  = (const float*)d_in[5];
  const float* reg_w  = (const float*)d_in[6];
  const float* reg_b  = (const float*)d_in[7];
  float* out = (float*)d_out;
  char* ws = (char*)d_ws;
  // workspace layout (bytes) -- ~29.1 MB total
  _Float16* featP = (_Float16*)(ws + 0);            // 10,653,696 (padded 102x102 NHWC)
  _Float16* w3h   = (_Float16*)(ws + 10653696);     //  4,718,592 (B-fragment order)
  _Float16* rpnH  = (_Float16*)(ws + 15372288);     // 10,240,000
  _Float16* w1h   = (_Float16*)(ws + 25612288);     //     49,152
  float*    scores= (float*)   (ws + 25661440);     //    360,000
  float*    prop  = (float*)   (ws + 26021440);     //  1,440,000
  float*    skey  = (float*)   (ws + 27461440);     //    524,288
  int*      sidx  = (int*)     (ws + 27985728);     //    524,288
  float*    p2    = (float*)   (ws + 28510016);     //     32,000
  unsigned* valid = (unsigned*)(ws + 28542016);     //      8,000
  unsigned* mask  = (unsigned*)(ws + 28550016);     //    504,000
  unsigned* keep  = (unsigned*)(ws + 29054016);     //      8,000
  int*      order = (int*)     (ws + 29062016);     //      8,000

  k_pack_feat<<<(PADW * PADW * CCH + 255) / 256, 256, 0, stream>>>(feat, featP);
  k_pack_w3  <<<(CCH * CCH * 9 + 255) / 256, 256, 0, stream>>>(w3, w3h);
  k_pack_w1  <<<(48 * CCH + 255) / 256, 256, 0, stream>>>(cls_w, reg_w, w1h);

  k_conv3<<<625, 256, 0, stream>>>(featP, w3h, b3, rpnH);
  k_conv1<<<625, 96, 0, stream>>>(rpnH, w1h, cls_b, reg_b, out, scores);

  k_anchors<<<(NPROP + 255) / 256, 256, 0, stream>>>(out);
  k_decode <<<(NPROP + 255) / 256, 256, 0, stream>>>(out, prop);

  k_sort_init<<<NSORT / 256, 256, 0, stream>>>(scores, skey, sidx);
  for (int k = 2; k <= NSORT; k <<= 1)
    for (int j2 = k >> 1; j2 > 0; j2 >>= 1)
      k_sort_pass<<<NSORT / 256, 256, 0, stream>>>(skey, sidx, k, j2);

  k_gather<<<(NPRE + 255) / 256, 256, 0, stream>>>(prop, sidx, p2, valid);
  k_iou   <<<NPRE, 64, 0, stream>>>(p2, mask);
  k_nms   <<<1, 64, 0, stream>>>(mask, valid, keep);
  k_final <<<1, 256, 0, stream>>>(keep, skey, p2, order, out);
}
